// CTAttention_9010841387717
// MI455X (gfx1250) — matrix operand order repaired and checked by reading, not compiled, not run on hardware
//
#include <hip/hip_runtime.h>
#include <hip/hip_bf16.h>

typedef _Float16 f16;
typedef __attribute__((ext_vector_type(16))) _Float16 v16h;
typedef __attribute__((ext_vector_type(8)))  _Float16 v8h;
typedef __attribute__((ext_vector_type(8)))  float    v8f;

// Problem constants (fixed in the reference module).
constexpr int TOTAL = 6464;
constexpr int LENS_[8]  = {1024, 896, 768, 640, 1024, 512, 900, 700};
constexpr int OFFS_[9]  = {0, 1024, 1920, 2688, 3328, 4352, 4864, 5764, 6464};
// prefix of ceil(L/32): {32,28,24,20,32,16,29,22}
constexpr int QTP32_[9] = {0, 32, 60, 84, 104, 136, 152, 181, 203};
constexpr int NQT32     = 203;
constexpr float SCALE   = 0.17677669529663687f;   // 32^-0.5
constexpr float NEGF    = -1e30f;

__device__ __forceinline__ v8f v8f_zero() {
  v8f z;
#pragma unroll
  for (int i = 0; i < 8; ++i) z[i] = 0.0f;
  return z;
}

__device__ __forceinline__ v8f wmma_f16(v16h a, v16h b, v8f c) {
  // D = A(16x32 f16) * B(32x16 f16) + C(16x16 f32)
  return __builtin_amdgcn_wmma_f32_16x16x32_f16(false, a, false, b, (short)0, c,
                                                false, false);
}

// v16h fragment from two 16-byte chunks at p and p+16 elements (A-operand shape).
__device__ __forceinline__ v16h load_frag_split16(const f16* p) {
  v8h lo = *(const v8h*)(p);
  v8h hi = *(const v8h*)(p + 16);
  v16h f;
#pragma unroll
  for (int j = 0; j < 8; ++j) { f[j] = lo[j]; f[8 + j] = hi[j]; }
  return f;
}

// v16h fragment from 16 contiguous f16 (contiguous k order, used for both A and B).
__device__ __forceinline__ v16h load_frag_contig(const f16* p) {
  v8h lo = *(const v8h*)(p);
  v8h hi = *(const v8h*)(p + 8);
  v16h f;
#pragma unroll
  for (int j = 0; j < 8; ++j) { f[j] = lo[j]; f[8 + j] = hi[j]; }
  return f;
}

__device__ __forceinline__ int batch_of(int t) {
  int b = 0;
#pragma unroll
  for (int i = 0; i < 7; ++i) if (t >= OFFS_[i + 1]) b = i + 1;
  return b;
}

// ---------------------------------------------------------------------------
// Kernel 0: f16-convert tokens and transpose + f16-convert the weights.
//   x16[t][k] = ct[t][k]
//   wT[c][k]  = qkv_w[k][c]  (768 x 256)
//   pT[c][k]  = proj_w[k][c] (256 x 256)
// ---------------------------------------------------------------------------
__global__ void prep_inputs(const float* __restrict__ ct,
                            const float* __restrict__ qkv_w,
                            const float* __restrict__ proj_w,
                            f16* __restrict__ x16, f16* __restrict__ wT,
                            f16* __restrict__ pT) {
  int idx = blockIdx.x * blockDim.x + threadIdx.x;
  if (idx < 768 * 256) {
    int c = idx >> 8, k = idx & 255;
    wT[idx] = (f16)qkv_w[k * 768 + c];
  }
  int i2 = idx - 768 * 256;
  if (i2 >= 0 && i2 < 256 * 256) {
    int c = i2 >> 8, k = i2 & 255;
    pT[i2] = (f16)proj_w[k * 256 + c];
  }
  int i3 = idx - (768 * 256 + 256 * 256);
  if (i3 >= 0 && i3 < TOTAL * 256) {
    x16[i3] = (f16)ct[i3];
  }
}

// ---------------------------------------------------------------------------
// Kernel 1: QKV projection.  x16 [TOTAL,256] f16 @ wT(f16) + bias.
// 4 waves per block; each wave computes a 16x16 output tile, K=256 in 8 WMMAs.
// Writes: q16 (scaled, [tok][256] f16), k16 ([tok][256] f16),
//         vt  (V transposed: [b][h*32+d][1024] f16).
// ---------------------------------------------------------------------------
__global__ __launch_bounds__(128) void qkv_project(
    const f16* __restrict__ x16, const float* __restrict__ qkv_b,
    const f16* __restrict__ wT, f16* __restrict__ q16, f16* __restrict__ k16,
    f16* __restrict__ vt) {
  const int lane = threadIdx.x & 31, wv = threadIdx.x >> 5;
  const int t0 = blockIdx.x * 16, c0 = (blockIdx.y * 4 + wv) * 16;
  const int n = lane & 15, hf = lane >> 4;

  const f16* ap = x16 + (size_t)(t0 + n) * 256;        // A row for this lane
  const f16* wp = wT + (size_t)(c0 + n) * 256;         // B column for this lane

  v8f acc = v8f_zero();
#pragma unroll
  for (int kc = 0; kc < 256; kc += 32) {
    v16h a  = load_frag_contig(ap + kc + hf * 16);
    v16h bf = load_frag_contig(wp + kc + hf * 16);
    acc = wmma_f16(a, bf, acc);
  }

  const int c = c0 + n;
  const float bias = qkv_b[c];
#pragma unroll
  for (int r = 0; r < 8; ++r) {
    const int t = t0 + r + 8 * hf;
    float val = acc[r] + bias;
    if (c < 256) {
      q16[(size_t)t * 256 + c] = (f16)(val * SCALE);
    } else if (c < 512) {
      k16[(size_t)t * 256 + (c - 256)] = (f16)val;
    } else {
      int b = batch_of(t);
      int w = t - OFFS_[b];
      vt[((size_t)b * 256 + (c - 512)) * 1024 + w] = (f16)val;
    }
  }
}

// ---------------------------------------------------------------------------
// Flash-softmax update for one 16-query group against a 32-key chunk.
// s0/s1 hold S^T tiles (lane: key m = r+8*hf, query n). Builds P^T via xor-16
// shuffles and accumulates O^T with two WMMAs.
// ---------------------------------------------------------------------------
__device__ __forceinline__ void flash_group(v8f s0, v8f s1, int kc, int L,
                                            int hf, float& m_run, float& l_run,
                                            v8f& o0, v8f& o1, v16h av0,
                                            v16h av1) {
  // Mask padded keys in the final chunk (element r = key kc + r + 8*hf).
#pragma unroll
  for (int r = 0; r < 8; ++r) {
    if (kc + r + 8 * hf >= L)      s0[r] = NEGF;
    if (kc + 16 + r + 8 * hf >= L) s1[r] = NEGF;
  }

  float cmax = NEGF;
#pragma unroll
  for (int r = 0; r < 8; ++r) {
    cmax = fmaxf(cmax, s0[r]);
    cmax = fmaxf(cmax, s1[r]);
  }
  cmax = fmaxf(cmax, __shfl_xor(cmax, 16, 32));

  const float m_new = fmaxf(m_run, cmax);
  const float alpha = __expf(m_run - m_new);

  v8f p0, p1;
  float csum = 0.0f;
#pragma unroll
  for (int r = 0; r < 8; ++r) {
    p0[r] = __expf(s0[r] - m_new);
    p1[r] = __expf(s1[r] - m_new);
    csum += p0[r] + p1[r];
  }
  csum += __shfl_xor(csum, 16, 32);
  l_run = l_run * alpha + csum;
  m_run = m_new;

  // B = P^T (32 keys x 16 queries) assembled from S^T fragments via xor-16.
  v16h pf;
#pragma unroll
  for (int r = 0; r < 8; ++r) {
    float x0 = __shfl_xor(p0[r], 16, 32);
    float x1 = __shfl_xor(p1[r], 16, 32);
    pf[r]     = (f16)(hf ? x1 : p0[r]);   // keys (hf?16:0)+0..7
    pf[8 + r] = (f16)(hf ? p1[r] : x0);   // keys (hf?16:0)+8..15
  }

#pragma unroll
  for (int r = 0; r < 8; ++r) { o0[r] *= alpha; o1[r] *= alpha; }
  o0 = wmma_f16(av0, pf, o0);
  o1 = wmma_f16(av1, pf, o1);
}

// ---------------------------------------------------------------------------
// Kernel 2: fused flash attention.  One wave per (batch, head, 32-query tile);
// 4 waves (4 heads) per workgroup.  S^T = K * Q^T so all loads are contiguous;
// both 16-query groups share the K/V A-fragments (halves K/V L2 traffic).
// ---------------------------------------------------------------------------
__global__ __launch_bounds__(128) void attn_flash(
    const f16* __restrict__ q16, const f16* __restrict__ k16,
    const f16* __restrict__ vt, f16* __restrict__ ao16) {
  const int lane = threadIdx.x & 31, wv = threadIdx.x >> 5;
  const int tile = blockIdx.x, h = blockIdx.y * 4 + wv;
  int b = 0;
#pragma unroll
  for (int i = 0; i < 7; ++i) if (tile >= QTP32_[i + 1]) b = i + 1;
  const int q0 = (tile - QTP32_[b]) * 32;
  const int off = OFFS_[b], L = LENS_[b];
  const int n = lane & 15, hf = lane >> 4;

  // Persistent B operands: Q^T for the two 16-query groups.
  int qa = q0 + n;      if (qa > L - 1) qa = L - 1;
  int qb = q0 + 16 + n; if (qb > L - 1) qb = L - 1;
  const v16h qfragA =
      load_frag_contig(q16 + (size_t)(off + qa) * 256 + h * 32 + hf * 16);
  const v16h qfragB =
      load_frag_contig(q16 + (size_t)(off + qb) * 256 + h * 32 + hf * 16);

  // A operands for O^T = V^T * P^T: rows are head-dims, contiguous in key.
  const f16* vb0 = vt + ((size_t)b * 256 + h * 32 + n) * 1024;
  const f16* vb1 = vt + ((size_t)b * 256 + h * 32 + 16 + n) * 1024;

  v8f o0a = v8f_zero(), o1a = v8f_zero(), o0b = v8f_zero(), o1b = v8f_zero();
  float mA = NEGF, lA = 0.0f, mB = NEGF, lB = 0.0f;

  for (int kc = 0; kc < L; kc += 32) {
    // S^T tiles: A = K rows (16 keys x 32 hd), shared by both query groups.
    int k0i = kc + n;      if (k0i > L - 1) k0i = L - 1;
    int k1i = kc + 16 + n; if (k1i > L - 1) k1i = L - 1;
    v16h a0 = load_frag_contig(k16 + (size_t)(off + k0i) * 256 + h * 32 + hf * 16);
    v16h a1 = load_frag_contig(k16 + (size_t)(off + k1i) * 256 + h * 32 + hf * 16);
    v8f s0a = wmma_f16(a0, qfragA, v8f_zero());
    v8f s1a = wmma_f16(a1, qfragA, v8f_zero());
    v8f s0b = wmma_f16(a0, qfragB, v8f_zero());
    v8f s1b = wmma_f16(a1, qfragB, v8f_zero());

    // V^T A-fragments, shared by both query groups.
    v16h av0 = load_frag_contig(vb0 + kc + hf * 16);
    v16h av1 = load_frag_contig(vb1 + kc + hf * 16);

    flash_group(s0a, s1a, kc, L, hf, mA, lA, o0a, o1a, av0, av1);
    flash_group(s0b, s1b, kc, L, hf, mB, lB, o0b, o1b, av0, av1);
  }

  const float invA = (lA > 0.0f) ? (1.0f / lA) : 0.0f;
  const float invB = (lB > 0.0f) ? (1.0f / lB) : 0.0f;
#pragma unroll
  for (int r = 0; r < 8; ++r) {
    const int d = r + 8 * hf;
    const int qA = q0 + n, qB = q0 + 16 + n;
    if (qA < L) {
      const size_t base = (size_t)(off + qA) * 256 + h * 32;
      ao16[base + d]      = (f16)(o0a[r] * invA);
      ao16[base + 16 + d] = (f16)(o1a[r] * invA);
    }
    if (qB < L) {
      const size_t base = (size_t)(off + qB) * 256 + h * 32;
      ao16[base + d]      = (f16)(o0b[r] * invB);
      ao16[base + 16 + d] = (f16)(o1b[r] * invB);
    }
  }
}

// ---------------------------------------------------------------------------
// Kernel 3: output projection.  ao16 [TOTAL,256] f16 @ pT + proj_b -> fp32.
// 4 waves per block, one 16x16 tile per wave.
// ---------------------------------------------------------------------------
__global__ __launch_bounds__(128) void out_project(
    const f16* __restrict__ ao16, const f16* __restrict__ pT,
    const float* __restrict__ proj_b, float* __restrict__ out) {
  const int lane = threadIdx.x & 31, wv = threadIdx.x >> 5;
  const int t0 = blockIdx.x * 16, c0 = (blockIdx.y * 4 + wv) * 16;
  const int n = lane & 15, hf = lane >> 4;

  const f16* ap = ao16 + (size_t)(t0 + n) * 256;
  const f16* bp = pT + (size_t)(c0 + n) * 256;

  v8f acc = v8f_zero();
#pragma unroll
  for (int kc = 0; kc < 256; kc += 32) {
    v16h a  = load_frag_contig(ap + kc + hf * 16);
    v16h bf = load_frag_contig(bp + kc + hf * 16);
    acc = wmma_f16(a, bf, acc);
  }

  const int c = c0 + n;
  const float bias = proj_b[c];
#pragma unroll
  for (int r = 0; r < 8; ++r) {
    const int t = t0 + r + 8 * hf;
    out[(size_t)t * 256 + c] = acc[r] + bias;
  }
}

// ---------------------------------------------------------------------------
extern "C" void kernel_launch(void* const* d_in, const int* in_sizes, int n_in,
                              void* d_out, int out_size, void* d_ws, size_t ws_size,
                              hipStream_t stream) {
  const float* ct     = (const float*)d_in[0];
  // d_in[1] = ct_mask  (unused: mask is implied by the ragged lengths)
  // d_in[2] = batch_num_windows (compile-time constants)
  const float* qkv_w  = (const float*)d_in[3];
  const float* qkv_b  = (const float*)d_in[4];
  const float* proj_w = (const float*)d_in[5];
  const float* proj_b = (const float*)d_in[6];
  float* out = (float*)d_out;

  // Workspace carve-up (256-byte aligned).
  char* ws = (char*)d_ws;
  size_t o = 0;
  auto alloc = [&](size_t bytes) {
    void* p = ws + o;
    o += (bytes + 255) & ~(size_t)255;
    return p;
  };
  f16* x16  = (f16*)alloc((size_t)TOTAL * 256 * sizeof(f16));
  f16* q16  = (f16*)alloc((size_t)TOTAL * 256 * sizeof(f16));
  f16* k16  = (f16*)alloc((size_t)TOTAL * 256 * sizeof(f16));
  f16* vt   = (f16*)alloc((size_t)8 * 256 * 1024 * sizeof(f16));
  f16* ao16 = (f16*)alloc((size_t)TOTAL * 256 * sizeof(f16));
  f16* wT   = (f16*)alloc((size_t)768 * 256 * sizeof(f16));
  f16* pT   = (f16*)alloc((size_t)256 * 256 * sizeof(f16));
  (void)ws_size; (void)in_sizes; (void)n_in; (void)out_size;

  // 0) input/weight f16 conversion (+ weight transpose)
  const int prep_elems = 768 * 256 + 256 * 256 + TOTAL * 256;
  prep_inputs<<<dim3((prep_elems + 255) / 256), dim3(256), 0, stream>>>(
      ct, qkv_w, proj_w, x16, wT, pT);
  // 1) QKV projection: 404 token tiles x 12 blocks x 4 waves (48 col tiles)
  qkv_project<<<dim3(TOTAL / 16, 12), dim3(128), 0, stream>>>(
      x16, qkv_b, wT, q16, k16, vt);
  // 2) fused flash attention: 203 query tiles x 2 blocks x 4 waves (8 heads)
  attn_flash<<<dim3(NQT32, 2), dim3(128), 0, stream>>>(q16, k16, vt, ao16);
  // 3) output projection: 404 token tiles x 4 blocks x 4 waves (16 col tiles)
  out_project<<<dim3(TOTAL / 16, 4), dim3(128), 0, stream>>>(ao16, pT, proj_b, out);
}
